// QuerySelector_46772193854058
// MI455X (gfx1250) — matrix operand order repaired and checked by reading, not compiled, not run on hardware
//
#include <hip/hip_runtime.h>
#include <hip/hip_bf16.h>
#include <cstdint>
#include <cstddef>

typedef __attribute__((ext_vector_type(16))) _Float16 v16h;
typedef __attribute__((ext_vector_type(4)))  _Float16 v4h;
typedef __attribute__((ext_vector_type(8)))  float    v8f;

#define B_SZ 16
#define L_SZ 2048
#define D_SZ 512
#define LQ   1372                 // int((1-0.33)*2048)
#define NQT  86                   // ceil(1372/16)
#define KB   128                  // keys per flash-attention iteration
#define QK_SCALE 0.04419417382415922f  // 1/sqrt(512)

// LDS byte offsets inside attn_kernel's dynamic shared block (base = 0, no static LDS)
#define SQ_OFF   0
#define SK_OFF   (16 * D_SZ * 2)                    // 16384   : K tile [128][512] f16
#define SV_OFF   (SK_OFF + KB * D_SZ * 2)           // +131072 : V tile [512][128] f16
#define SS_OFF   (SV_OFF + D_SZ * KB * 2)           // +131072 : S [16][128] f32
#define SP_OFF   (SS_OFF + 16 * KB * 4)             // P [16][128] f16
#define STAT_OFF (SP_OFF + 16 * KB * 2)
#define SMEM_TOTAL (STAT_OFF + 16 * 4 * 3 + 256 * 4)   // ~292 KB (<320 KB WGP LDS)

// ---------------------------------------------------------------------------
// Kernel 0a: keys f32 -> f16 (row-major), elementwise
// ---------------------------------------------------------------------------
__global__ __launch_bounds__(256)
void kcvt_kernel(const float* __restrict__ src, _Float16* __restrict__ dst) {
  const size_t gid = (size_t)blockIdx.x * 256 + threadIdx.x;  // float4 index
  const float4 v = *(const float4*)&src[gid * 4];
  v4h h; h[0] = (_Float16)v.x; h[1] = (_Float16)v.y; h[2] = (_Float16)v.z; h[3] = (_Float16)v.w;
  *(v4h*)&dst[gid * 4] = h;
}

// ---------------------------------------------------------------------------
// Kernel 0b: values f32 -> f16 TRANSPOSED: vT[b, d, l]  (32x32 LDS tiles)
// ---------------------------------------------------------------------------
__global__ __launch_bounds__(256)
void vcvtT_kernel(const float* __restrict__ values, _Float16* __restrict__ vT) {
  __shared__ _Float16 sT[32][33];
  const int b  = blockIdx.x >> 10;         // 64 * 16 tiles per batch
  const int lt = (blockIdx.x >> 4) & 63;
  const int dt = blockIdx.x & 15;
  const int l0 = lt * 32, d0 = dt * 32;
  const int t = threadIdx.x;
  for (int e = t; e < 1024; e += 256) {
    const int li = e >> 5, di = e & 31;
    sT[li][di] = (_Float16)values[((size_t)b * L_SZ + l0 + li) * D_SZ + d0 + di];
  }
  __syncthreads();
  for (int e = t; e < 1024; e += 256) {
    const int di = e >> 5, li = e & 31;
    vT[((size_t)b * D_SZ + d0 + di) * L_SZ + l0 + li] = sT[li][di];
  }
}

// ---------------------------------------------------------------------------
// Kernel 1: K_reduce[b,d] = mean of top-1372 of keys[b,:,d]  (bitonic sort)
// ---------------------------------------------------------------------------
__global__ __launch_bounds__(256)
void kred_kernel(const float* __restrict__ keys, float* __restrict__ kred) {
  __shared__ float sv[L_SZ];
  __shared__ float red[256];
  const int t = threadIdx.x;
  const int b = blockIdx.x >> 9;
  const int d = blockIdx.x & 511;
  for (int i = t; i < L_SZ; i += 256)
    sv[i] = keys[((size_t)b * L_SZ + i) * D_SZ + d];
  __syncthreads();
  for (int k = 2; k <= L_SZ; k <<= 1) {
    for (int j = k >> 1; j > 0; j >>= 1) {
      for (int i = t; i < L_SZ; i += 256) {
        const int ixj = i ^ j;
        if (ixj > i) {
          const float v1 = sv[i], v2 = sv[ixj];
          const bool inorder = ((i & k) == 0);      // descending blocks
          const bool doSwap = inorder ? (v1 < v2) : (v1 > v2);
          if (doSwap) { sv[i] = v2; sv[ixj] = v1; }
        }
      }
      __syncthreads();
    }
  }
  float p = 0.f;
  for (int i = t; i < LQ; i += 256) p += sv[i];
  red[t] = p;
  __syncthreads();
  for (int s = 128; s > 0; s >>= 1) {
    if (t < s) red[t] += red[t + s];
    __syncthreads();
  }
  if (t == 0) kred[b * D_SZ + d] = red[0] * (1.0f / (float)LQ);
}

// ---------------------------------------------------------------------------
// Kernel 2: sqk[b,l] = sum_d kred[b,d] * queries[b,l,d]   (wave per row)
// ---------------------------------------------------------------------------
__global__ __launch_bounds__(256)
void sqk_kernel(const float* __restrict__ queries, const float* __restrict__ kred,
                float* __restrict__ sqk) {
  const int lane = threadIdx.x & 31;
  const int w    = threadIdx.x >> 5;
  const int row  = blockIdx.x * 8 + w;          // [0, B*L)
  const int b = row >> 11;
  const int l = row & 2047;
  const size_t qbase = ((size_t)b * L_SZ + l) * D_SZ;
  float acc = 0.f;
  #pragma unroll
  for (int i = 0; i < 16; ++i)
    acc += queries[qbase + lane + 32 * i] * kred[b * D_SZ + lane + 32 * i];
  #pragma unroll
  for (int off = 16; off > 0; off >>= 1)
    acc += __shfl_down(acc, off, 32);
  if (lane == 0) sqk[b * L_SZ + l] = acc;
}

// ---------------------------------------------------------------------------
// Kernel 3: per-batch top-k of sqk -> sorted query indices (bitonic KV sort)
// ---------------------------------------------------------------------------
__global__ __launch_bounds__(256)
void topk_kernel(const float* __restrict__ sqk, int* __restrict__ idxBuf) {
  __shared__ float sv[L_SZ];
  __shared__ int   si[L_SZ];
  const int t = threadIdx.x;
  const int b = blockIdx.x;
  for (int i = t; i < L_SZ; i += 256) { sv[i] = sqk[b * L_SZ + i]; si[i] = i; }
  __syncthreads();
  for (int k = 2; k <= L_SZ; k <<= 1) {
    for (int j = k >> 1; j > 0; j >>= 1) {
      for (int i = t; i < L_SZ; i += 256) {
        const int ixj = i ^ j;
        if (ixj > i) {
          const float v1 = sv[i], v2 = sv[ixj];
          const int   i1 = si[i], i2 = si[ixj];
          const bool before12 = (v1 > v2) || (v1 == v2 && i1 < i2);
          const bool inorder  = ((i & k) == 0);
          const bool doSwap   = inorder ? !before12 : before12;
          if (doSwap) { sv[i] = v2; sv[ixj] = v1; si[i] = i2; si[ixj] = i1; }
        }
      }
      __syncthreads();
    }
  }
  for (int i = t; i < L_SZ; i += 256) idxBuf[b * L_SZ + i] = si[i];
}

// ---------------------------------------------------------------------------
// Kernel 4: meanV[b,d] = mean_l values[b,l,d]
// ---------------------------------------------------------------------------
__global__ __launch_bounds__(256)
void meanv_kernel(const float* __restrict__ values, float* __restrict__ meanV) {
  const int b = blockIdx.x >> 1;
  const int d = ((blockIdx.x & 1) << 8) + threadIdx.x;
  float acc = 0.f;
  for (int l = 0; l < L_SZ; ++l)
    acc += values[((size_t)b * L_SZ + l) * D_SZ + d];
  meanV[b * D_SZ + d] = acc * (1.0f / (float)L_SZ);
}

// ---------------------------------------------------------------------------
// Kernel 5: fill out[b,l,d] = meanV[b,d]  (float4)
// ---------------------------------------------------------------------------
__global__ __launch_bounds__(256)
void fill_kernel(const float* __restrict__ meanV, float* __restrict__ out) {
  const int gid = blockIdx.x * 256 + threadIdx.x;   // float4 index
  const int b   = gid >> 18;
  const int d4  = gid & 127;
  const float4 mv = *(const float4*)&meanV[b * D_SZ + d4 * 4];
  *(float4*)&out[(size_t)gid * 4] = mv;
}

// ---------------------------------------------------------------------------
// Kernel 6: flash attention over selected queries.
//   Software-pipelined GLOBAL_LOAD_ASYNC_TO_LDS_B128 staging (ASYNCcnt is
//   in-order, so s_wait_asynccnt 32 retires exactly the older tile),
//   v_wmma_f32_16x16x32_f16 for both QK^T and PV.
// ---------------------------------------------------------------------------
__device__ __forceinline__ void async_lds_b128(unsigned lds_off, uint64_t gaddr) {
  asm volatile("global_load_async_to_lds_b128 %0, %1, off"
               :: "v"(lds_off), "v"(gaddr) : "memory");
}
__device__ __forceinline__ void wait_async_le32() {   // older 32 of <=64 done
  asm volatile("s_wait_asynccnt 0x20" ::: "memory");
}
__device__ __forceinline__ void wait_async0() {
  asm volatile("s_wait_asynccnt 0x0" ::: "memory");
}

// issue 32 async 16B chunks/thread: K block (row-major) -> sK
__device__ __forceinline__ void stage_k(uint64_t kBatch, int kb, int t) {
  const uint64_t gbase = kBatch + (uint64_t)kb * D_SZ * 2;
  for (int e = t; e < KB * (D_SZ / 8); e += 256) {
    const int kr = e >> 6, d8 = e & 63;
    async_lds_b128(SK_OFF + (unsigned)(kr * D_SZ + d8 * 8) * 2,
                   gbase + ((uint64_t)kr * D_SZ + d8 * 8) * 2);
  }
}
// issue 32 async 16B chunks/thread: V block (d-major) -> sV as [512][128]
__device__ __forceinline__ void stage_v(uint64_t vBatch, int kb, int t) {
  const uint64_t gbase = vBatch + (uint64_t)kb * 2;
  for (int e = t; e < D_SZ * (KB / 8); e += 256) {
    const int d = e >> 4, k8 = e & 15;
    async_lds_b128(SV_OFF + (unsigned)(d * KB + k8 * 8) * 2,
                   gbase + ((uint64_t)d * L_SZ + k8 * 8) * 2);
  }
}

__global__ __launch_bounds__(256)
void attn_kernel(const float* __restrict__ queries,
                 const _Float16* __restrict__ kf16,   // [B, L, D] f16
                 const _Float16* __restrict__ vT,     // [B, D, L] f16 (transposed)
                 const int*   __restrict__ idxBuf,
                 float*       __restrict__ out) {
  extern __shared__ __align__(16) char smem[];
  _Float16* sQ  = (_Float16*)(smem + SQ_OFF);     // 16 x 512 f16
  _Float16* sK  = (_Float16*)(smem + SK_OFF);     // [128][512] f16
  _Float16* sV  = (_Float16*)(smem + SV_OFF);     // [512][128] f16
  float*    sS  = (float*)   (smem + SS_OFF);     // 16 x 128 f32
  _Float16* sP  = (_Float16*)(smem + SP_OFF);     // 16 x 128 f16
  float* sRowMax = (float*)(smem + STAT_OFF);     // 16
  float* sRowSum = sRowMax + 16;                  // 16
  float* sCorr   = sRowSum + 16;                  // 16
  float* sRed    = sCorr + 16;                    // 256

  const int t    = threadIdx.x;
  const int w    = t >> 5;
  const int lane = t & 31;
  const int half = lane >> 4;
  const int mrow = lane & 15;
  const int b    = blockIdx.x / NQT;
  const int qt   = blockIdx.x % NQT;

  const uint64_t kBatch = (uint64_t)(uintptr_t)(kf16 + (size_t)b * L_SZ * D_SZ);
  const uint64_t vBatch = (uint64_t)(uintptr_t)(vT   + (size_t)b * D_SZ * L_SZ);

  // ---- prologue: start K(0) streaming while we gather Q ----
  stage_k(kBatch, 0, t);

  // ---- load gathered Q tile (f32 -> f16) ----
  for (int e = t; e < 16 * (D_SZ / 4); e += 256) {
    const int r = e >> 7, d4 = e & 127;
    const int qi = qt * 16 + r;
    const int qc = qi < LQ ? qi : LQ - 1;
    const int qidx = idxBuf[b * L_SZ + qc];
    const float4 v = *(const float4*)&queries[((size_t)b * L_SZ + qidx) * D_SZ + d4 * 4];
    _Float16* dst = &sQ[r * D_SZ + d4 * 4];
    dst[0] = (_Float16)v.x; dst[1] = (_Float16)v.y;
    dst[2] = (_Float16)v.z; dst[3] = (_Float16)v.w;
  }
  if (t < 16) { sRowMax[t] = -1e30f; sRowSum[t] = 0.f; }

  v8f oacc[4];
  #pragma unroll
  for (int j = 0; j < 4; ++j)
    #pragma unroll
    for (int r = 0; r < 8; ++r) oacc[j][r] = 0.f;

  for (int kb = 0; kb < L_SZ; kb += KB) {
    const bool notLast = (kb + KB) < L_SZ;

    // start V(i) streaming; it overlaps the S-phase WMMAs below
    stage_v(vBatch, kb, t);
    wait_async_le32();          // in-order: retires the older 32 = K(i)
    __syncthreads();            // all waves see K(i) in sK (and sQ on iter 0)

    // ---- S = Q * K^T (this wave's 16 keys, full D contraction) ----
    v8f sc;
    #pragma unroll
    for (int r = 0; r < 8; ++r) sc[r] = 0.f;
    const int krowBase = (16 * w + mrow) * D_SZ;
    #pragma unroll 4
    for (int k0 = 0; k0 < D_SZ; k0 += 32) {
      v16h a, bf;
      const int qbase = mrow * D_SZ + k0 + half * 16;
      #pragma unroll
      for (int i = 0; i < 16; ++i) a[i] = sQ[qbase + i];
      const int kbase = krowBase + k0 + half * 16;
      #pragma unroll
      for (int i = 0; i < 16; ++i) bf[i] = sK[kbase + i];
      sc = __builtin_amdgcn_wmma_f32_16x16x32_f16(false, a, false, bf,
                                                  (short)0, sc, false, false);
    }
    #pragma unroll
    for (int r = 0; r < 8; ++r)
      sS[(r + 8 * half) * KB + 16 * w + mrow] = sc[r] * QK_SCALE;
    __syncthreads();            // S published; sK free

    // start K(i+1) streaming; it overlaps softmax + PV below
    if (notLast) stage_k(kBatch, kb + KB, t);

    // ---- online softmax (16 rows x 128 cols; 16 threads per row) ----
    const int row = t >> 4, sub = t & 15;
    float lm = -1e30f;
    #pragma unroll
    for (int j = 0; j < 8; ++j) lm = fmaxf(lm, sS[row * KB + sub * 8 + j]);
    sRed[t] = lm;
    __syncthreads();
    if (t < 16) {
      float bm = sRed[t * 16];
      for (int i = 1; i < 16; ++i) bm = fmaxf(bm, sRed[t * 16 + i]);
      const float mo = sRowMax[t];
      const float mn = fmaxf(mo, bm);
      const float corr = __expf(mo - mn);
      sCorr[t] = corr;
      sRowMax[t] = mn;
      sRowSum[t] *= corr;
    }
    __syncthreads();
    const float mrv = sRowMax[row];
    float ps = 0.f;
    #pragma unroll
    for (int j = 0; j < 8; ++j) {
      const float p = __expf(sS[row * KB + sub * 8 + j] - mrv);
      sP[row * KB + sub * 8 + j] = (_Float16)p;
      ps += p;
    }
    sRed[t] = ps;
    __syncthreads();
    if (t < 16) {
      float s = 0.f;
      for (int i = 0; i < 16; ++i) s += sRed[t * 16 + i];
      sRowSum[t] += s;
    }
    // rescale O accumulators by per-row correction
    #pragma unroll
    for (int r = 0; r < 8; ++r) {
      const float cr = sCorr[r + 8 * half];
      #pragma unroll
      for (int j = 0; j < 4; ++j) oacc[j][r] *= cr;
    }

    // V(i) done? outstanding = V(i) [older] + K(i+1) [newer]
    if (notLast) wait_async_le32(); else wait_async0();
    __syncthreads();            // all waves see V(i) in sV; sP published

    // ---- O += P * V   (wave's 64-wide d-slice; contiguous B-fragments) ----
    #pragma unroll
    for (int j = 0; j < 4; ++j) {
      const int dcol = 64 * w + 16 * j;
      #pragma unroll
      for (int k0 = 0; k0 < KB; k0 += 32) {
        v16h a, bf;
        const int pbase = mrow * KB + k0 + half * 16;
        #pragma unroll
        for (int i = 0; i < 16; ++i) a[i] = sP[pbase + i];
        const int vbase = (dcol + mrow) * KB + k0 + half * 16;
        #pragma unroll
        for (int i = 0; i < 16; ++i) bf[i] = sV[vbase + i];
        oacc[j] = __builtin_amdgcn_wmma_f32_16x16x32_f16(false, a, false, bf,
                                                         (short)0, oacc[j], false, false);
      }
    }
    __syncthreads();            // sV and sP free for next iteration
  }

  // ---- finalize: O / rowSum, scatter to selected query rows ----
  #pragma unroll
  for (int r = 0; r < 8; ++r) {
    const int ql = r + 8 * half;
    const int qi = qt * 16 + ql;
    if (qi < LQ) {
      const int qidx = idxBuf[b * L_SZ + qi];
      const float invl = 1.0f / sRowSum[ql];
      #pragma unroll
      for (int j = 0; j < 4; ++j)
        out[((size_t)b * L_SZ + qidx) * D_SZ + 64 * w + 16 * j + mrow] = oacc[j][r] * invl;
    }
  }
}

// ---------------------------------------------------------------------------
extern "C" void kernel_launch(void* const* d_in, const int* in_sizes, int n_in,
                              void* d_out, int out_size, void* d_ws, size_t ws_size,
                              hipStream_t stream) {
  const float* queries = (const float*)d_in[0];
  const float* keys    = (const float*)d_in[1];
  const float* values  = (const float*)d_in[2];
  float* out = (float*)d_out;

  // workspace layout
  float* kred  = (float*)d_ws;                    // B*512 f32
  float* sqk   = kred + B_SZ * D_SZ;              // B*2048 f32
  int*   idxB  = (int*)(sqk + B_SZ * L_SZ);       // B*2048 i32
  float* meanV = (float*)(idxB + B_SZ * L_SZ);    // B*512 f32
  _Float16* kf16 = (_Float16*)(meanV + B_SZ * D_SZ);        // B*L*D f16 (32 MB)
  _Float16* vT   = kf16 + (size_t)B_SZ * L_SZ * D_SZ;       // B*D*L f16 (32 MB)

  (void)hipFuncSetAttribute((const void*)attn_kernel,
                            hipFuncAttributeMaxDynamicSharedMemorySize,
                            (int)SMEM_TOTAL);

  const int nElem4 = B_SZ * L_SZ * D_SZ / 4;      // float4 count

  kcvt_kernel <<<nElem4 / 256,               256, 0, stream>>>(keys, kf16);
  vcvtT_kernel<<<B_SZ * 64 * 16,             256, 0, stream>>>(values, vT);
  kred_kernel <<<B_SZ * D_SZ,                256, 0, stream>>>(keys, kred);
  sqk_kernel  <<<(B_SZ * L_SZ) / 8,          256, 0, stream>>>(queries, kred, sqk);
  topk_kernel <<<B_SZ,                       256, 0, stream>>>(sqk, idxB);
  meanv_kernel<<<B_SZ * 2,                   256, 0, stream>>>(values, meanV);
  fill_kernel <<<nElem4 / 256,               256, 0, stream>>>(meanV, out);
  attn_kernel <<<B_SZ * NQT, 256, SMEM_TOTAL, stream>>>(queries, kf16, vT, idxB, out);
}
